// HOIPositionNetV5_23785528885550
// MI455X (gfx1250) — compile-verified, matrix operand order checked by reading, not run on hardware
//
#include <hip/hip_runtime.h>
#include <hip/hip_fp16.h>

// ---------------------------------------------------------------------------
// HOIPositionNet on MI455X (gfx1250, wave32, WMMA f16->f32).
//
// Restructured: mean_tok[n] = subj[n] + bsaf + ((S_h2[n]/cnt)@wi3 + bi3 + pe1)@w_i
//                             + Sobj[n]/cnt   (diag fallback analogous)
// so only the first two pairwise-MLP layers run over all 256x256 pairs
// (~43 GFLOP f16 WMMA); wi3 / w_i matmuls collapse to 256 rows.
//
// Pair kernel: 512 threads (16 waves), m-chunks of 128 -> each wi2^T fragment
// is reused across 8 row-tiles, halving L2 weight traffic vs 64-row chunks
// (aggregate L2 demand ~30 TB/s, under a plausible L2 ceiling; HBM untouched).
// ---------------------------------------------------------------------------

typedef __attribute__((ext_vector_type(16))) _Float16 v16h;
typedef __attribute__((ext_vector_type(8)))  _Float16 v8h;
typedef __attribute__((ext_vector_type(8)))  float    v8f;

#define DEVINL __device__ __forceinline__

// Load one WMMA 16-bit operand fragment (A layout; B fed as B^T rows in the
// same layout): per lane two contiguous 16-byte chunks, K-halves split by
// lane half per ISA 7.12.2 (lanes 0-15: K {k0..k0+7, k0+16..k0+23}, lanes
// 16-31: {k0+8..k0+15, k0+24..k0+31}); caller pre-offsets by (lane>>4)*8.
DEVINL v16h load_frag(const _Float16* p) {
    v8h lo = *(const v8h*)(p);
    v8h hi = *(const v8h*)(p + 16);
    v16h r;
#pragma unroll
    for (int i = 0; i < 8; ++i) { r[i] = lo[i]; r[i + 8] = hi[i]; }
    return r;
}

DEVINL v8f wmma16(v16h a, v16h b, v8f c) {
    return __builtin_amdgcn_wmma_f32_16x16x32_f16(false, a, false, b, (short)0, c, false, false);
}

DEVINL float siluf(float x) { return x / (1.f + __expf(-x)); }

// ------------------------------- prep kernels ------------------------------

__global__ __launch_bounds__(256) void bbox_kernel(const float* __restrict__ poly,
                                                   float* __restrict__ bbox) {
    int o = threadIdx.x;
    if (o < 256) {
        float x1 = 1e30f, y1 = 1e30f, x2 = -1e30f, y2 = -1e30f;
#pragma unroll
        for (int j = 0; j < 4; ++j) {
            float x = poly[o * 8 + 2 * j], y = poly[o * 8 + 2 * j + 1];
            x1 = fminf(x1, x); x2 = fmaxf(x2, x);
            y1 = fminf(y1, y); y2 = fmaxf(y2, y);
        }
        bbox[o * 4 + 0] = x1; bbox[o * 4 + 1] = y1;
        bbox[o * 4 + 2] = x2; bbox[o * 4 + 3] = y2;
    }
}

// WT[n][k] = (f16) W[k][n]   (W is K x N row-major)
__global__ __launch_bounds__(256) void transposeh_kernel(const float* __restrict__ W,
                                                         _Float16* __restrict__ WT,
                                                         int K, int N) {
    int idx = blockIdx.x * 256 + threadIdx.x;
    if (idx < K * N) {
        int n = idx / K, k = idx % K;
        WT[idx] = (_Float16)W[k * N + n];
    }
}

// Fourier features of the 256 polygons -> f16 (256 x 128)
__global__ __launch_bounds__(256) void fourier_poly_kernel(const float* __restrict__ poly,
                                                           _Float16* __restrict__ Xe) {
    int idx = blockIdx.x * 256 + threadIdx.x;
    if (idx < 256 * 128) {
        int o = idx >> 7, c = idx & 127;
        int f = c >> 4, j = c & 15;
        float freq = __powf(100.f, (float)f * 0.125f);
        float x = poly[o * 8 + (j & 7)];
        float v = (j < 8) ? __sinf(x * freq) : __cosf(x * freq);
        Xe[idx] = (_Float16)v;
    }
}

// As = f16(ent + pe0), Ao = f16(ent + pe2)
__global__ __launch_bounds__(256) void addpe_kernel(const float* __restrict__ ent,
                                                    const float* __restrict__ pe,
                                                    _Float16* __restrict__ As,
                                                    _Float16* __restrict__ Ao) {
    int idx = blockIdx.x * 256 + threadIdx.x;
    if (idx < 256 * 768) {
        int col = idx % 768;
        float e = ent[idx];
        As[idx] = (_Float16)(e + pe[col]);
        Ao[idx] = (_Float16)(e + pe[1536 + col]);
    }
}

// ------------------------- generic WMMA tile GEMM --------------------------
// C[M][N] = act(A[M][K] @ B + bias + add2); B provided transposed (BT = N x K).
// One 16x16 output tile per wave; A/B streamed from global (L2-resident).
__global__ __launch_bounds__(128) void gemm_tile_kernel(
    const _Float16* __restrict__ A, const _Float16* __restrict__ BT,
    const float* __restrict__ bias, const float* __restrict__ add2,
    int M, int N, int K, int act,
    _Float16* __restrict__ Ch, float* __restrict__ Cf) {
    const int lane = threadIdx.x & 31, wave = threadIdx.x >> 5;
    const int ntN = N >> 4;
    const int tile = blockIdx.x * 4 + wave;
    if (tile >= (M >> 4) * ntN) return;
    const int mt = tile / ntN, nt = tile % ntN;
    const int l15 = lane & 15, half = lane >> 4;
    const _Float16* ap = A + (size_t)(mt * 16 + l15) * K + half * 8;
    const _Float16* bp = BT + (size_t)(nt * 16 + l15) * K + half * 8;
    v8f acc = {};
    for (int kb = 0; kb < K; kb += 32)
        acc = wmma16(load_frag(ap + kb), load_frag(bp + kb), acc);
    const int col = nt * 16 + l15;
    const float bb = (bias ? bias[col] : 0.f) + (add2 ? add2[col] : 0.f);
#pragma unroll
    for (int r = 0; r < 8; ++r) {
        const int row = mt * 16 + r + half * 8;   // C/D layout: M = r + 8*(lane>=16)
        float v = acc[r] + bb;
        if (act) v = siluf(v);
        if (Ch) Ch[(size_t)row * N + col] = (_Float16)v;
        if (Cf) Cf[(size_t)row * N + col] = v;
    }
}

// --------------------------- heavy pairwise kernel -------------------------
// One workgroup (16 waves, 512 threads) per subject n; 2 chunks of 128 m.
// Per chunk: fourier -> LDS (128x128 f16), layer1 WMMA -> LDS H1 (128x512 f16),
// layer2 WMMA -> silu -> vf-weighted sum into S_h2[n][512], diag capture.
// Each wave owns 2 output-column tiles with 8 row-tile accumulators, so each
// streamed weight fragment is reused 8x.
__global__ __launch_bounds__(512) void pair_kernel(
    const float* __restrict__ bbox,
    const _Float16* __restrict__ wiT1,   // [512][128] = wi1^T
    const _Float16* __restrict__ wiT2,   // [512][512] = wi2^T
    const float* __restrict__ bi1, const float* __restrict__ bi2,
    float* __restrict__ S_h2, float* __restrict__ diag_h2,
    float* __restrict__ cnt_out) {
    __shared__ float    sb[1024];
    __shared__ _Float16 sX[128 * 136];   // fourier features (row stride 136, 16B aligned)
    __shared__ _Float16 sH[128 * 528];   // hidden1 (row stride 528, 16B aligned)
    __shared__ float    sS[512];
    __shared__ float    sD[512];
    __shared__ float    sWgt[128];
    __shared__ float    sCnt;

    const int tid = threadIdx.x;
    const int lane = tid & 31, wave = tid >> 5;       // 16 waves
    const int l15 = lane & 15, half = lane >> 4;
    const int n = blockIdx.x;

    for (int i = tid; i < 1024; i += 512) sb[i] = bbox[i];
    if (tid < 512) { sS[tid] = 0.f; sD[tid] = 0.f; }
    if (tid == 0) sCnt = 0.f;
    __syncthreads();

    const float nx1 = sb[n * 4 + 0], ny1 = sb[n * 4 + 1];
    const float nx2 = sb[n * 4 + 2], ny2 = sb[n * 4 + 3];

    for (int chunk = 0; chunk < 2; ++chunk) {
        const int mbase = chunk * 128;
        // ---- fourier features + validity weights ----
        {
            const int r = tid >> 2, g = tid & 3;      // r: 0..127, g: 0..3
            const int m = mbase + r;
            const float ix1 = fmaxf(nx1, sb[m * 4 + 0]), iy1 = fmaxf(ny1, sb[m * 4 + 1]);
            const float ix2 = fminf(nx2, sb[m * 4 + 2]), iy2 = fminf(ny2, sb[m * 4 + 3]);
            const float c8[8] = {ix1, iy1, ix2, iy1, ix2, iy2, ix1, iy2};
#pragma unroll
            for (int ff = 0; ff < 2; ++ff) {
                const int f = g * 2 + ff;
                const float freq = __powf(100.f, (float)f * 0.125f);
#pragma unroll
                for (int j = 0; j < 8; ++j) {
                    const float a = c8[j] * freq;
                    sX[r * 136 + f * 16 + j]     = (_Float16)__sinf(a);
                    sX[r * 136 + f * 16 + 8 + j] = (_Float16)__cosf(a);
                }
            }
            if (g == 0)
                sWgt[r] = ((ix2 > ix1) && (iy2 > iy1) && (m != n)) ? 1.f : 0.f;
        }
        __syncthreads();
        if (tid == 0) {   // deterministic count accumulation
            float a = 0.f;
            for (int r = 0; r < 128; ++r) a += sWgt[r];
            sCnt += a;
        }

        // ---- layer 1: 128 -> 512, silu, into sH ----
        for (int dti = 0; dti < 2; ++dti) {
            const int dt = wave * 2 + dti;            // 0..31
            const int d = dt * 16 + l15;
            v8f acc[8] = {{}, {}, {}, {}, {}, {}, {}, {}};
            const _Float16* bp = wiT1 + (size_t)d * 128 + half * 8;
            const _Float16* a0 = &sX[l15 * 136 + half * 8];
            for (int kb = 0; kb < 128; kb += 32) {
                const v16h b = load_frag(bp + kb);
#pragma unroll
                for (int mt = 0; mt < 8; ++mt)
                    acc[mt] = wmma16(load_frag(a0 + mt * 16 * 136 + kb), b, acc[mt]);
            }
            const float bia = bi1[d];
#pragma unroll
            for (int mt = 0; mt < 8; ++mt)
#pragma unroll
                for (int r = 0; r < 8; ++r) {
                    const int row = mt * 16 + r + half * 8;
                    sH[row * 528 + d] = (_Float16)siluf(acc[mt][r] + bia);
                }
        }
        __syncthreads();

        // ---- layer 2: 512 -> 512, silu, weighted reduce over m ----
        for (int dti = 0; dti < 2; ++dti) {
            const int dt = wave * 2 + dti;
            const int d = dt * 16 + l15;
            v8f acc[8] = {{}, {}, {}, {}, {}, {}, {}, {}};
            const _Float16* bp = wiT2 + (size_t)d * 512 + half * 8;
            const _Float16* a0 = &sH[l15 * 528 + half * 8];
            for (int kb = 0; kb < 512; kb += 32) {
                const v16h b = load_frag(bp + kb);
#pragma unroll
                for (int mt = 0; mt < 8; ++mt)
                    acc[mt] = wmma16(load_frag(a0 + mt * 16 * 528 + kb), b, acc[mt]);
            }
            const float bia = bi2[d];
            float part = 0.f;
#pragma unroll
            for (int mt = 0; mt < 8; ++mt)
#pragma unroll
                for (int r = 0; r < 8; ++r) {
                    const int mrow = mt * 16 + r + half * 8;
                    const float s = siluf(acc[mt][r] + bia);
                    part += s * sWgt[mrow];
                    if (mbase + mrow == n) sD[d] = s;   // single writer
                }
            // combine the two lane-halves (same d), single deterministic writer
            part += __shfl_down(part, 16);
            if (half == 0) sS[d] += part;   // wave owns d exclusively: no race
        }
        __syncthreads();
    }

    if (tid < 512) {
        S_h2[(size_t)n * 512 + tid]    = sS[tid];
        diag_h2[(size_t)n * 512 + tid] = sD[tid];
    }
    if (tid == 0) cnt_out[n] = sCnt;
}

// ------------------------------ tail kernels -------------------------------

__global__ __launch_bounds__(256) void makez_kernel(const float* __restrict__ S,
                                                    const float* __restrict__ D,
                                                    const float* __restrict__ cnt,
                                                    _Float16* __restrict__ Z) {
    int idx = blockIdx.x * 256 + threadIdx.x;
    if (idx < 256 * 512) {
        float c = cnt[idx >> 9];
        Z[idx] = (_Float16)((c > 0.f) ? S[idx] / c : D[idx]);
    }
}

__global__ __launch_bounds__(256) void final_kernel(
    const float* __restrict__ bbox, const float* __restrict__ cnt,
    const float* __restrict__ subj, const float* __restrict__ obj,
    const float* __restrict__ P, const float* __restrict__ bsaf,
    float* __restrict__ out) {
    __shared__ float fb[1024];
    const int n = blockIdx.x, t = threadIdx.x;
    for (int i = t; i < 1024; i += 256) fb[i] = bbox[i];
    __syncthreads();
    const float c = cnt[n];
    const bool has = c > 0.f;
    float o0, o1, o2;
    if (has) {
        const float nx1 = fb[n * 4], ny1 = fb[n * 4 + 1];
        const float nx2 = fb[n * 4 + 2], ny2 = fb[n * 4 + 3];
        float a0 = 0.f, a1 = 0.f, a2 = 0.f;
        for (int m = 0; m < 256; ++m) {
            const float ix1 = fmaxf(nx1, fb[m * 4]), iy1 = fmaxf(ny1, fb[m * 4 + 1]);
            const float ix2 = fminf(nx2, fb[m * 4 + 2]), iy2 = fminf(ny2, fb[m * 4 + 3]);
            if ((ix2 > ix1) && (iy2 > iy1) && (m != n)) {
                a0 += obj[m * 768 + t];
                a1 += obj[m * 768 + t + 256];
                a2 += obj[m * 768 + t + 512];
            }
        }
        o0 = a0 / c; o1 = a1 / c; o2 = a2 / c;
    } else {
        o0 = obj[n * 768 + t];
        o1 = obj[n * 768 + t + 256];
        o2 = obj[n * 768 + t + 512];
    }
    const float oc[3] = {o0, o1, o2};
#pragma unroll
    for (int cc = 0; cc < 3; ++cc) {
        const int col = t + cc * 256;
        const size_t id = (size_t)n * 768 + col;
        out[id] = subj[id] + bsaf[col] + P[id] + oc[cc];
    }
}

// ------------------------------ host orchestration -------------------------

static constexpr size_t alup(size_t x) { return (x + 255) & ~(size_t)255; }

static constexpr size_t OFF_BBOX = 0;
static constexpr size_t OFF_WET1 = alup(OFF_BBOX + 256 * 4 * 4);
static constexpr size_t OFF_WET2 = alup(OFF_WET1 + 512 * 128 * 2);
static constexpr size_t OFF_WET3 = alup(OFF_WET2 + 512 * 512 * 2);
static constexpr size_t OFF_WIT1 = alup(OFF_WET3 + 768 * 512 * 2);
static constexpr size_t OFF_WIT2 = alup(OFF_WIT1 + 512 * 128 * 2);
static constexpr size_t OFF_WIT3 = alup(OFF_WIT2 + 512 * 512 * 2);
static constexpr size_t OFF_WST  = alup(OFF_WIT3 + 768 * 512 * 2);
static constexpr size_t OFF_WIS  = alup(OFF_WST + 768 * 768 * 2);
static constexpr size_t OFF_WOT  = alup(OFF_WIS + 768 * 768 * 2);
static constexpr size_t OFF_XE   = alup(OFF_WOT + 768 * 768 * 2);
static constexpr size_t OFF_H1E  = alup(OFF_XE + 256 * 128 * 2);
static constexpr size_t OFF_H2E  = alup(OFF_H1E + 256 * 512 * 2);
static constexpr size_t OFF_ENT  = alup(OFF_H2E + 256 * 512 * 2);
static constexpr size_t OFF_AS   = alup(OFF_ENT + 256 * 768 * 4);
static constexpr size_t OFF_AO   = alup(OFF_AS + 256 * 768 * 2);
static constexpr size_t OFF_SUBJ = alup(OFF_AO + 256 * 768 * 2);
static constexpr size_t OFF_OBJ  = alup(OFF_SUBJ + 256 * 768 * 4);
static constexpr size_t OFF_SH2  = alup(OFF_OBJ + 256 * 768 * 4);
static constexpr size_t OFF_DH2  = alup(OFF_SH2 + 256 * 512 * 4);
static constexpr size_t OFF_CNT  = alup(OFF_DH2 + 256 * 512 * 4);
static constexpr size_t OFF_Z    = alup(OFF_CNT + 256 * 4);
static constexpr size_t OFF_UH   = alup(OFF_Z + 256 * 512 * 2);
static constexpr size_t OFF_P    = alup(OFF_UH + 256 * 768 * 2);

extern "C" void kernel_launch(void* const* d_in, const int* in_sizes, int n_in,
                              void* d_out, int out_size, void* d_ws, size_t ws_size,
                              hipStream_t stream) {
    (void)in_sizes; (void)n_in; (void)out_size; (void)ws_size;
    const float* poly = (const float*)d_in[0];
    const float* pe   = (const float*)d_in[1];
    const float* we1  = (const float*)d_in[2];  const float* be1 = (const float*)d_in[3];
    const float* we2  = (const float*)d_in[4];  const float* be2 = (const float*)d_in[5];
    const float* we3  = (const float*)d_in[6];  const float* be3 = (const float*)d_in[7];
    const float* wi1  = (const float*)d_in[8];  const float* bi1 = (const float*)d_in[9];
    const float* wi2  = (const float*)d_in[10]; const float* bi2 = (const float*)d_in[11];
    const float* wi3  = (const float*)d_in[12]; const float* bi3 = (const float*)d_in[13];
    const float* wsaf = (const float*)d_in[14]; const float* bsaf = (const float*)d_in[15];
    float* out = (float*)d_out;

    char* ws = (char*)d_ws;
    float*    bbox = (float*)(ws + OFF_BBOX);
    _Float16* weT1 = (_Float16*)(ws + OFF_WET1);
    _Float16* weT2 = (_Float16*)(ws + OFF_WET2);
    _Float16* weT3 = (_Float16*)(ws + OFF_WET3);
    _Float16* wiT1 = (_Float16*)(ws + OFF_WIT1);
    _Float16* wiT2 = (_Float16*)(ws + OFF_WIT2);
    _Float16* wiT3 = (_Float16*)(ws + OFF_WIT3);
    _Float16* wsT  = (_Float16*)(ws + OFF_WST);
    _Float16* wiS  = (_Float16*)(ws + OFF_WIS);
    _Float16* woT  = (_Float16*)(ws + OFF_WOT);
    _Float16* Xe   = (_Float16*)(ws + OFF_XE);
    _Float16* H1e  = (_Float16*)(ws + OFF_H1E);
    _Float16* H2e  = (_Float16*)(ws + OFF_H2E);
    float*    entf = (float*)(ws + OFF_ENT);
    _Float16* As   = (_Float16*)(ws + OFF_AS);
    _Float16* Ao   = (_Float16*)(ws + OFF_AO);
    float*    subj = (float*)(ws + OFF_SUBJ);
    float*    objp = (float*)(ws + OFF_OBJ);
    float*    Sh2  = (float*)(ws + OFF_SH2);
    float*    Dh2  = (float*)(ws + OFF_DH2);
    float*    cntb = (float*)(ws + OFF_CNT);
    _Float16* Z    = (_Float16*)(ws + OFF_Z);
    _Float16* Uh   = (_Float16*)(ws + OFF_UH);
    float*    P    = (float*)(ws + OFF_P);

    bbox_kernel<<<1, 256, 0, stream>>>(poly, bbox);

    struct TJob { const float* W; _Float16* WT; int K, N; };
    const TJob jobs[9] = {
        {we1, weT1, 128, 512}, {we2, weT2, 512, 512}, {we3, weT3, 512, 768},
        {wi1, wiT1, 128, 512}, {wi2, wiT2, 512, 512}, {wi3, wiT3, 512, 768},
        {wsaf,                 wsT, 768, 768},
        {wsaf + 768 * 768,     wiS, 768, 768},
        {wsaf + 2 * 768 * 768, woT, 768, 768},
    };
    for (int i = 0; i < 9; ++i) {
        int total = jobs[i].K * jobs[i].N;
        transposeh_kernel<<<(total + 255) / 256, 256, 0, stream>>>(jobs[i].W, jobs[i].WT,
                                                                   jobs[i].K, jobs[i].N);
    }

    fourier_poly_kernel<<<128, 256, 0, stream>>>(poly, Xe);

    auto gemm = [&](const _Float16* A, const _Float16* BT, const float* bias,
                    const float* add2, int M, int N, int K, int act,
                    _Float16* Ch, float* Cf) {
        int tiles = (M / 16) * (N / 16);
        gemm_tile_kernel<<<tiles / 4, 128, 0, stream>>>(A, BT, bias, add2, M, N, K, act, Ch, Cf);
    };

    // entity MLP + projections
    gemm(Xe,  weT1, be1, nullptr, 256, 512, 128, 1, H1e, nullptr);
    gemm(H1e, weT2, be2, nullptr, 256, 512, 512, 1, H2e, nullptr);
    gemm(H2e, weT3, be3, nullptr, 256, 768, 512, 0, nullptr, entf);
    addpe_kernel<<<768, 256, 0, stream>>>(entf, pe, As, Ao);
    gemm(As, wsT, nullptr, nullptr, 256, 768, 768, 0, nullptr, subj);
    gemm(Ao, woT, nullptr, nullptr, 256, 768, 768, 0, nullptr, objp);

    // heavy pairwise MLP (layers 1-2) with fused masked reduction
    pair_kernel<<<256, 512, 0, stream>>>(bbox, wiT1, wiT2, bi1, bi2, Sh2, Dh2, cntb);

    // collapsed tail: z -> (z@wi3 + bi3 + pe1) -> @w_i -> combine
    makez_kernel<<<512, 256, 0, stream>>>(Sh2, Dh2, cntb, Z);
    gemm(Z,  wiT3, bi3, pe + 768, 256, 768, 512, 0, Uh, nullptr);
    gemm(Uh, wiS, nullptr, nullptr, 256, 768, 768, 0, nullptr, P);
    final_kernel<<<256, 256, 0, stream>>>(bbox, cntb, subj, objp, P, bsaf, out);
}
